// RGCN_3891240370438
// MI455X (gfx1250) — compile-verified, matrix operand order checked by reading, not run on hardware
//
#include <hip/hip_runtime.h>
#include <hip/hip_bf16.h>

typedef __bf16 bf16_t;
typedef __attribute__((ext_vector_type(16))) __bf16 v16bf;
typedef __attribute__((ext_vector_type(8)))  __bf16 v8bf;
typedef __attribute__((ext_vector_type(8)))  float  v8f;

#define N_NODES 100000
#define R_REL   8
#define DIM     128
#define N_BASES 6
#define N_EDGES 1600000
#define KTOT    1152              // R*D (relations) + D (root/self term)
#define KCHUNKS 36                // KTOT / 32
#define NTILES  (N_NODES / 16)    // 6250 tiles of 16 nodes

// ---------------------------------------------------------------------------
// Zero-fill (sums + cnt), float4 granularity
// ---------------------------------------------------------------------------
__global__ void rgcn_zero_f4(float4* __restrict__ p, long n4) {
    long i = (long)blockIdx.x * blockDim.x + threadIdx.x;
    if (i < n4) p[i] = make_float4(0.f, 0.f, 0.f, 0.f);
}

// ---------------------------------------------------------------------------
// Build W[r,d,o] = sum_b comp[r,b]*basis[b,d,o]  (k<1024), root (k>=1024),
// and store bf16 in the exact per-lane WMMA B-operand layout:
//   element i of lane L's v16bf for chunk kc, col-tile oc:
//     col n = 16*oc + (L&15); kbase = kc*32 + ((L<16)?0:8)
//     i<8  -> k = kbase + i ; i>=8 -> k = kbase + 16 + (i-8)
// ---------------------------------------------------------------------------
__global__ void rgcn_pack_weights(const float* __restrict__ comp,
                                  const float* __restrict__ basis,
                                  const float* __restrict__ root,
                                  bf16_t* __restrict__ Bpack) {
    int idx = blockIdx.x * blockDim.x + threadIdx.x;
    if (idx >= KTOT * DIM) return;
    int k = idx >> 7;         // 0..1151
    int o = idx & 127;
    float val;
    if (k < R_REL * DIM) {
        int r = k >> 7, d = k & 127;
        float acc = 0.f;
#pragma unroll
        for (int b = 0; b < N_BASES; ++b)
            acc += comp[r * N_BASES + b] * basis[(b * DIM + d) * DIM + o];
        val = acc;
    } else {
        val = root[(k - R_REL * DIM) * DIM + o];
    }
    int kc      = k >> 5;
    int kr      = k & 31;
    int laneGrp = (kr >> 3) & 1;                 // 0: lanes 0-15, 1: lanes 16-31
    int i       = (kr & 7) | ((kr & 16) >> 1);   // element slot 0..15
    int lane    = laneGrp * 16 + (o & 15);
    int oc      = o >> 4;
    Bpack[((((kc * 8 + oc) << 5) + lane) << 4) + i] = (bf16_t)val;
}

// ---------------------------------------------------------------------------
// Edge scatter: one wave (32 lanes) per edge, float4 per lane,
// f32 atomics into sums[(dst*R+rel)*D + :] and cnt[dst*R+rel].
// ---------------------------------------------------------------------------
__global__ void rgcn_scatter(const float* __restrict__ x,
                             const int*  __restrict__ src,
                             const int*  __restrict__ dst,
                             const int*  __restrict__ etype,
                             float* __restrict__ sums,
                             float* __restrict__ cnt) {
    int t = blockIdx.x * blockDim.x + threadIdx.x;
    int e = t >> 5;
    if (e >= N_EDGES) return;
    int lane = t & 31;
    int s = src[e], d = dst[e], r = etype[e];
    int seg = d * R_REL + r;
    const float4 v = ((const float4*)(x + (size_t)s * DIM))[lane];
    float* base = sums + (size_t)seg * DIM + lane * 4;
    atomicAdd(base + 0, v.x);
    atomicAdd(base + 1, v.y);
    atomicAdd(base + 2, v.z);
    atomicAdd(base + 3, v.w);
    if (lane == 0) atomicAdd(cnt + seg, 1.0f);
}

// ---------------------------------------------------------------------------
// Fused mean + relation GEMM + root GEMM.
// Block = 256 threads (8 waves) owns 16 output rows (nodes).
// Phase 1: stage A[16][1152] in LDS as bf16 (sums * 1/max(cnt,1); x appended).
// Phase 2: wave w owns output columns [16w,16w+16); 36x v_wmma_f32_16x16x32_bf16.
// ---------------------------------------------------------------------------
__global__ __launch_bounds__(256)
void rgcn_matmul(const float*  __restrict__ sums,
                 const float*  __restrict__ cnt,
                 const float*  __restrict__ xin,
                 const bf16_t* __restrict__ Bpack,
                 float* __restrict__ out) {
    __shared__ bf16_t Asm[16 * KTOT];   // 36864 B
    __shared__ float  rcs[16 * R_REL];

    int tid = threadIdx.x;
    int n0  = blockIdx.x * 16;

    if (tid < 16 * R_REL) {
        int nn = tid >> 3, rr = tid & 7;
        float c = cnt[(size_t)(n0 + nn) * R_REL + rr];
        rcs[tid] = 1.0f / fmaxf(c, 1.0f);
    }
    __syncthreads();

    for (int idx = tid; idx < 16 * KTOT; idx += 256) {
        int m = idx / KTOT;
        int k = idx - m * KTOT;
        float v;
        if (k < R_REL * DIM) {
            int r = k >> 7, d = k & 127;
            v = sums[((size_t)(n0 + m) * R_REL + r) * DIM + d] * rcs[(m << 3) + r];
        } else {
            v = xin[(size_t)(n0 + m) * DIM + (k - R_REL * DIM)];
        }
        Asm[idx] = (bf16_t)v;
    }
    __syncthreads();

    int wave = tid >> 5, lane = tid & 31;
    int mrow = lane & 15;
    int kofs = (lane & 16) >> 1;   // 0 (lanes 0-15) or 8 (lanes 16-31)

    v8f acc = {};
    const bf16_t* bp = Bpack + (((size_t)wave << 5) + lane) * 16;

    for (int kc = 0; kc < KCHUNKS; ++kc) {
        const bf16_t* ap = &Asm[mrow * KTOT + kc * 32 + kofs];
        v8bf a0 = *(const v8bf*)ap;          // K = kbase .. kbase+7
        v8bf a1 = *(const v8bf*)(ap + 16);   // K = kbase+16 .. kbase+23
        v16bf a = __builtin_shufflevector(a0, a1,
                    0, 1, 2, 3, 4, 5, 6, 7, 8, 9, 10, 11, 12, 13, 14, 15);
        v16bf b = *(const v16bf*)(bp + (size_t)kc * 8 * 32 * 16);
        acc = __builtin_amdgcn_wmma_f32_16x16x32_bf16(
                  /*neg_a=*/false, a, /*neg_b=*/false, b,
                  /*c_mod=*/(short)0, acc, /*reuse_a=*/false, /*reuse_b=*/false);
    }

    int col  = (wave << 4) + (lane & 15);
    int rofs = (lane & 16) >> 1;   // rows 0-7 or 8-15
#pragma unroll
    for (int j = 0; j < 8; ++j)
        out[(size_t)(n0 + rofs + j) * DIM + col] = acc[j];
}

// ---------------------------------------------------------------------------
// LayerNorm (+optional ReLU, +optional residual), one wave per node row,
// wave32 shuffle reductions. In-place on io.
// ---------------------------------------------------------------------------
__global__ void rgcn_layernorm(float* __restrict__ io,
                               const float* __restrict__ g,
                               const float* __restrict__ b,
                               const float* __restrict__ resid,
                               const float* __restrict__ res_scale_p,
                               int do_relu) {
    int t    = blockIdx.x * blockDim.x + threadIdx.x;
    int node = t >> 5;
    if (node >= N_NODES) return;
    int lane = t & 31;

    float4 v = ((const float4*)(io + (size_t)node * DIM))[lane];
    float s = v.x + v.y + v.z + v.w;
    float q = v.x * v.x + v.y * v.y + v.z * v.z + v.w * v.w;
#pragma unroll
    for (int o = 16; o > 0; o >>= 1) {
        s += __shfl_xor(s, o, 32);
        q += __shfl_xor(q, o, 32);
    }
    float mean = s * (1.0f / DIM);
    float var  = q * (1.0f / DIM) - mean * mean;
    float rstd = rsqrtf(var + 1e-5f);

    float4 gv = ((const float4*)g)[lane];
    float4 bv = ((const float4*)b)[lane];
    float4 y;
    y.x = (v.x - mean) * rstd * gv.x + bv.x;
    y.y = (v.y - mean) * rstd * gv.y + bv.y;
    y.z = (v.z - mean) * rstd * gv.z + bv.z;
    y.w = (v.w - mean) * rstd * gv.w + bv.w;
    if (do_relu) {
        y.x = fmaxf(y.x, 0.f); y.y = fmaxf(y.y, 0.f);
        y.z = fmaxf(y.z, 0.f); y.w = fmaxf(y.w, 0.f);
    }
    if (resid) {
        float rs = *res_scale_p;
        float4 h = ((const float4*)(resid + (size_t)node * DIM))[lane];
        y.x += rs * h.x; y.y += rs * h.y; y.z += rs * h.z; y.w += rs * h.w;
    }
    ((float4*)(io + (size_t)node * DIM))[lane] = y;
}

// ---------------------------------------------------------------------------
extern "C" void kernel_launch(void* const* d_in, const int* in_sizes, int n_in,
                              void* d_out, int out_size, void* d_ws, size_t ws_size,
                              hipStream_t stream) {
    const int*   edge_index = (const int*)d_in[0];
    const int*   src        = edge_index;
    const int*   dst        = edge_index + N_EDGES;
    const int*   etype      = (const int*)d_in[1];
    const float* emb        = (const float*)d_in[2];
    const float* comp1      = (const float*)d_in[3];
    const float* basis1     = (const float*)d_in[4];
    const float* root1      = (const float*)d_in[5];
    const float* gamma1     = (const float*)d_in[6];
    const float* beta1      = (const float*)d_in[7];
    const float* comp2      = (const float*)d_in[8];
    const float* basis2     = (const float*)d_in[9];
    const float* root2      = (const float*)d_in[10];
    const float* gamma2     = (const float*)d_in[11];
    const float* beta2      = (const float*)d_in[12];
    const float* res_scale  = (const float*)d_in[13];

    // Workspace layout
    const size_t SUMS_ELEMS  = (size_t)N_NODES * R_REL * DIM;   // 102.4M f32
    const size_t CNT_ELEMS   = (size_t)N_NODES * R_REL;         // 0.8M f32
    const size_t BPACK_ELEMS = (size_t)KTOT * DIM;              // 147456 bf16
    char* ws = (char*)d_ws;
    float*  sums = (float*)ws;
    float*  cnt  = sums + SUMS_ELEMS;
    bf16_t* Bp1  = (bf16_t*)(ws + (SUMS_ELEMS + CNT_ELEMS) * sizeof(float));
    bf16_t* Bp2  = Bp1 + BPACK_ELEMS;
    float*  h    = (float*)((char*)Bp2 + BPACK_ELEMS * sizeof(bf16_t));
    float*  outp = (float*)d_out;

    dim3 blk(256);
    long zero4   = (long)(SUMS_ELEMS + CNT_ELEMS) / 4;
    int  zgrid   = (int)((zero4 + 255) / 256);
    int  pgrid   = (KTOT * DIM + 255) / 256;
    int  sgrid   = (int)(((size_t)N_EDGES * 32 + 255) / 256);
    int  lngrid  = (N_NODES * 32 + 255) / 256;

    // Pack both layers' weights into WMMA-friendly bf16 B operands
    rgcn_pack_weights<<<pgrid, blk, 0, stream>>>(comp1, basis1, root1, Bp1);
    rgcn_pack_weights<<<pgrid, blk, 0, stream>>>(comp2, basis2, root2, Bp2);

    // ---- Layer 1 ----
    rgcn_zero_f4<<<zgrid, blk, 0, stream>>>((float4*)sums, zero4);
    rgcn_scatter<<<sgrid, blk, 0, stream>>>(emb, src, dst, etype, sums, cnt);
    rgcn_matmul<<<NTILES, blk, 0, stream>>>(sums, cnt, emb, Bp1, h);
    rgcn_layernorm<<<lngrid, blk, 0, stream>>>(h, gamma1, beta1,
                                               nullptr, nullptr, /*relu=*/1);

    // ---- Layer 2 ----
    rgcn_zero_f4<<<zgrid, blk, 0, stream>>>((float4*)sums, zero4);
    rgcn_scatter<<<sgrid, blk, 0, stream>>>(h, src, dst, etype, sums, cnt);
    rgcn_matmul<<<NTILES, blk, 0, stream>>>(sums, cnt, h, Bp2, outp);
    rgcn_layernorm<<<lngrid, blk, 0, stream>>>(outp, gamma2, beta2,
                                               h, res_scale, /*relu=*/0);
}
